// SwinTransformerBlock_16097537425691
// MI455X (gfx1250) — matrix operand order repaired and checked by reading, not compiled, not run on hardware
//
#include <hip/hip_runtime.h>
#include <hip/hip_bf16.h>
#include <math.h>

// ---------------------------------------------------------------------------
// Swin Transformer block for MI455X (gfx1250), wave32 + WMMA f16 path.
// Dense GEMMs: 64x64 register-blocked macro-tile per wave -> 16 independent
// v_wmma_f32_16x16x32_f16 per k-step with 4x fragment reuse. Attention uses
// LDS for the P-matrix re-layout and V^T staging. Activations live in the
// 192MB L2 between phases, so fragments load straight from global.
// ---------------------------------------------------------------------------

typedef _Float16 v16h __attribute__((ext_vector_type(16)));
typedef _Float16 v8h  __attribute__((ext_vector_type(8)));
typedef float    v8f  __attribute__((ext_vector_type(8)));

#define C_DIM    192
#define HID_DIM  768
#define NHEADS   6
#define HD       32
#define WSZ      8
#define NWTOK    64           // tokens per window
#define SHIFT_SZ 4
#define IMG      128
#define BATCH    8
#define TOK      (BATCH*IMG*IMG)   // 131072
#define BWIN     (TOK/NWTOK)       // 2048
#define QKV_DIM  (3*C_DIM)         // 576

#define CAT16(a,b) __builtin_shufflevector((a),(b),0,1,2,3,4,5,6,7,8,9,10,11,12,13,14,15)

__device__ __forceinline__ float wave_sum32(float x) {
#pragma unroll
  for (int m = 16; m > 0; m >>= 1) x += __shfl_xor(x, m, 32);
  return x;
}
__device__ __forceinline__ float half_max16(float x) {
#pragma unroll
  for (int m = 8; m > 0; m >>= 1) x = fmaxf(x, __shfl_xor(x, m, 32));
  return x;
}
__device__ __forceinline__ float half_sum16(float x) {
#pragma unroll
  for (int m = 8; m > 0; m >>= 1) x += __shfl_xor(x, m, 32);
  return x;
}
// shift-window mask region id: [0,120)=0, [120,124)=1, [124,128)=2
__device__ __forceinline__ int region_of(int p) {
  return (p < IMG - WSZ) ? 0 : ((p < IMG - SHIFT_SZ) ? 1 : 2);
}

// --------------------------- f32 -> f16 weight cvt -------------------------
__global__ void cvt_f32_f16_kernel(const float* __restrict__ s,
                                   _Float16* __restrict__ d, int n) {
  int i = blockIdx.x * 256 + threadIdx.x;
  if (i < n) d[i] = (_Float16)s[i];
}

// --------------------------- LayerNorm (C=192) -----------------------------
// 4 waves / block, one token row per wave; lane handles channels c = i*32+lane.
// shifted==1: destination row is a window-partitioned token of roll(x, -4, -4).
__global__ void __launch_bounds__(128)
ln_kernel(const float* __restrict__ xin, const float* __restrict__ g,
          const float* __restrict__ bta, _Float16* __restrict__ outh,
          int shifted) {
  const int rw   = blockIdx.x * 4 + (threadIdx.x >> 5);
  const int lane = threadIdx.x & 31;
  int src = rw;
  if (shifted) {
    int b_ = rw >> 6, n = rw & 63;
    int b  = b_ >> 8, w = b_ & 255;
    int wh = w >> 4,  ww = w & 15;
    int i  = n >> 3,  j  = n & 7;
    int ys = ((wh << 3) + i + SHIFT_SZ) & (IMG - 1);
    int xs = ((ww << 3) + j + SHIFT_SZ) & (IMG - 1);
    src = (b * IMG + ys) * IMG + xs;
  }
  const float* px = xin + (size_t)src * C_DIM;
  float v[6]; float s = 0.f;
#pragma unroll
  for (int i = 0; i < 6; ++i) { v[i] = px[i * 32 + lane]; s += v[i]; }
  s = wave_sum32(s);
  const float mu = s * (1.f / C_DIM);
  float var = 0.f;
#pragma unroll
  for (int i = 0; i < 6; ++i) { float d = v[i] - mu; var += d * d; }
  var = wave_sum32(var) * (1.f / C_DIM);
  const float rstd = rsqrtf(var + 1e-5f);
  _Float16* po = outh + (size_t)rw * C_DIM;
#pragma unroll
  for (int i = 0; i < 6; ++i) {
    int c = i * 32 + lane;
    po[c] = (_Float16)((v[i] - mu) * rstd * g[c] + bta[c]);
  }
}

// --------------------------- blocked WMMA GEMM -----------------------------
// C[M,N] = A[M,K](f16 rowmajor) * W[N,K](f16 rowmajor)^T, f32 accumulate.
// One 64x64 macro-tile per wave (4x4 blocking): per k-step load 4 A-frags +
// 4 B-frags, issue 16 independent WMMAs. 8 waves/block. Epilogues:
//  0: +bias -> f16            (qkv)
//  1: +bias, window-reverse + roll(+4) scatter, +shortcut x -> f32 x1 (proj)
//  2: +bias, exact GELU -> f16 (fc1)
//  3: +bias, +resid -> f32     (fc2 -> final output)
template <int EPI>
__global__ void __launch_bounds__(256)
gemm64_kernel(const _Float16* __restrict__ A, const _Float16* __restrict__ W,
              const float* __restrict__ bias, _Float16* __restrict__ outH,
              float* __restrict__ outF, const float* __restrict__ resid,
              int M, int N, int K) {
  const int Nt4  = N >> 6;                       // 64-wide macro tiles
  const int wid  = blockIdx.x * 8 + (threadIdx.x >> 5);
  const int tm4  = wid / Nt4;
  const int tn4  = wid % Nt4;
  const int lane = threadIdx.x & 31;
  const int lo   = lane & 15;
  const int hi   = lane >> 4;

  const _Float16* arow[4];
  const _Float16* brow[4];
#pragma unroll
  for (int bm = 0; bm < 4; ++bm)
    arow[bm] = A + (size_t)(tm4 * 64 + bm * 16 + lo) * K;
#pragma unroll
  for (int bn = 0; bn < 4; ++bn)
    brow[bn] = W + (size_t)(tn4 * 64 + bn * 16 + lo) * K + hi * 16;

  v8f acc[4][4];
#pragma unroll
  for (int bm = 0; bm < 4; ++bm)
#pragma unroll
    for (int bn = 0; bn < 4; ++bn) acc[bm][bn] = (v8f){};

  for (int k0 = 0; k0 < K; k0 += 32) {
    v16h av[4], bv[4];
#pragma unroll
    for (int bm = 0; bm < 4; ++bm) {
      v8h a0 = *(const v8h*)(arow[bm] + k0 + hi * 16);
      v8h a1 = *(const v8h*)(arow[bm] + k0 + hi * 16 + 8);
      av[bm] = CAT16(a0, a1);
    }
#pragma unroll
    for (int bn = 0; bn < 4; ++bn) bv[bn] = *(const v16h*)(brow[bn] + k0);
#pragma unroll
    for (int bm = 0; bm < 4; ++bm)
#pragma unroll
      for (int bn = 0; bn < 4; ++bn)
        acc[bm][bn] = __builtin_amdgcn_wmma_f32_16x16x32_f16(
            false, av[bm], false, bv[bn], (short)0, acc[bm][bn], false, false);
  }

#pragma unroll
  for (int bn = 0; bn < 4; ++bn) {
    const int col  = tn4 * 64 + bn * 16 + lo;
    const float bc = bias[col];
#pragma unroll
    for (int bm = 0; bm < 4; ++bm) {
#pragma unroll
      for (int r = 0; r < 8; ++r) {
        const int row = tm4 * 64 + bm * 16 + r + hi * 8;
        float v = acc[bm][bn][r] + bc;
        if constexpr (EPI == 0) {
          outH[(size_t)row * N + col] = (_Float16)v;
        } else if constexpr (EPI == 1) {
          int b_ = row >> 6, n = row & 63;
          int b  = b_ >> 8, w = b_ & 255;
          int wh = w >> 4,  ww = w & 15;
          int i  = n >> 3,  j  = n & 7;
          int y  = ((wh << 3) + i + SHIFT_SZ) & (IMG - 1);
          int x  = ((ww << 3) + j + SHIFT_SZ) & (IMG - 1);
          size_t dtok = ((size_t)b * IMG + y) * IMG + x;
          outF[dtok * C_DIM + col] = v + resid[dtok * C_DIM + col];
        } else if constexpr (EPI == 2) {
          float gl = 0.5f * v * (1.0f + erff(v * 0.70710678118654752f));
          outH[(size_t)row * N + col] = (_Float16)gl;
        } else {
          outF[(size_t)row * N + col] = v + resid[(size_t)row * N + col];
        }
      }
    }
  }
}

// --------------------------- fused window attention ------------------------
// One block = one (window, head). 4 waves; wave tm owns query rows
// tm*16..tm*16+15. QK^T: 4 WMMAs/wave (K=hd=32 is one k-step). Softmax rows
// live entirely in one wave (16-lane half reductions). P routed via LDS to
// re-layout into A-fragments; V^T staged to LDS for contiguous B-fragments.
__global__ void __launch_bounds__(128)
swin_attn_kernel(const _Float16* __restrict__ qkv, const float* __restrict__ rpb,
                 _Float16* __restrict__ outA) {
  const int bwin = blockIdx.x / NHEADS;
  const int h    = blockIdx.x % NHEADS;
  const int tm   = threadIdx.x >> 5;
  const int lane = threadIdx.x & 31;
  const int lo   = lane & 15;
  const int hi   = lane >> 4;

  __shared__ alignas(64) _Float16 Pl[4][16][72];   // softmax(P), padded rows
  __shared__ alignas(64) _Float16 vT[32][80];      // V^T, row stride 160B

  // ---- stage V^T: 128 threads x 16 contiguous halves ----
  {
    int t = threadIdx.x;
    int key = t >> 1;
    int dbase = (t & 1) * 16;
    const _Float16* pv =
        qkv + (size_t)(bwin * NWTOK + key) * QKV_DIM + 2 * C_DIM + h * HD + dbase;
    v16h vv = *(const v16h*)pv;
#pragma unroll
    for (int i = 0; i < 16; ++i) vT[dbase + i][key] = vv[i];
  }
  __syncthreads();

  // ---- Q fragment (rows tm*16..tm*16+15, K = hd = 32) ----
  const _Float16* qrow =
      qkv + (size_t)(bwin * NWTOK + tm * 16 + lo) * QKV_DIM + h * HD;
  v8h q0 = *(const v8h*)(qrow + hi * 16);
  v8h q1 = *(const v8h*)(qrow + hi * 16 + 8);
  v16h qa = CAT16(q0, q1);

  // ---- S = Q K^T : 4 column tiles ----
  v8f acc[4];
#pragma unroll
  for (int tn = 0; tn < 4; ++tn) {
    const _Float16* krow = qkv + (size_t)(bwin * NWTOK + tn * 16 + lo) * QKV_DIM +
                           C_DIM + h * HD + hi * 16;
    v16h kb = *(const v16h*)krow;
    v8f z = {};
    acc[tn] = __builtin_amdgcn_wmma_f32_16x16x32_f16(false, qa, false, kb,
                                                     (short)0, z, false, false);
  }

  // ---- scale + relative-position bias + shift mask ----
  const float scale = 0.17677669529663687f;  // 1/sqrt(32)
  const int w  = bwin & 255;
  const int wh = w >> 4, ww = w & 15;
#pragma unroll
  for (int tn = 0; tn < 4; ++tn) {
    const int kidx = tn * 16 + lo;
    const int ky = kidx >> 3, kx = kidx & 7;
    const int lk = region_of(wh * 8 + ky) * 3 + region_of(ww * 8 + kx);
#pragma unroll
    for (int r = 0; r < 8; ++r) {
      const int qidx = tm * 16 + r + hi * 8;
      const int qy = qidx >> 3, qx = qidx & 7;
      const int lq = region_of(wh * 8 + qy) * 3 + region_of(ww * 8 + qx);
      const int rel = (qy - ky + 7) * 15 + (qx - kx + 7);
      const float bv = rpb[rel * NHEADS + h];
      const float mv = (lq != lk) ? -100.0f : 0.0f;
      acc[tn][r] = acc[tn][r] * scale + bv + mv;
    }
  }

  // ---- softmax along keys (64) and store P -> LDS ----
#pragma unroll
  for (int r = 0; r < 8; ++r) {
    float mx = fmaxf(fmaxf(acc[0][r], acc[1][r]), fmaxf(acc[2][r], acc[3][r]));
    mx = half_max16(mx);
    float p[4]; float s = 0.f;
#pragma unroll
    for (int tn = 0; tn < 4; ++tn) { p[tn] = expf(acc[tn][r] - mx); s += p[tn]; }
    s = half_sum16(s);
    const float inv = 1.0f / s;
    const int prow = r + hi * 8;
#pragma unroll
    for (int tn = 0; tn < 4; ++tn)
      Pl[tm][prow][tn * 16 + lo] = (_Float16)(p[tn] * inv);
  }
  __syncthreads();

  // ---- O = P V : K = 64 (2 k-steps), 2 output column tiles (hd=32) ----
#pragma unroll
  for (int tn2 = 0; tn2 < 2; ++tn2) {
    v8f oa = {};
#pragma unroll
    for (int k0 = 0; k0 < 64; k0 += 32) {
      v8h p0 = *(const v8h*)(&Pl[tm][lo][k0 + hi * 16]);
      v8h p1 = *(const v8h*)(&Pl[tm][lo][k0 + hi * 16 + 8]);
      v16h pa = CAT16(p0, p1);
      v16h vb = *(const v16h*)(&vT[tn2 * 16 + lo][k0 + hi * 16]);
      oa = __builtin_amdgcn_wmma_f32_16x16x32_f16(false, pa, false, vb,
                                                  (short)0, oa, false, false);
    }
#pragma unroll
    for (int r = 0; r < 8; ++r) {
      const int row = bwin * NWTOK + tm * 16 + r + hi * 8;
      const int col = h * HD + tn2 * 16 + lo;
      outA[(size_t)row * C_DIM + col] = (_Float16)oa[r];
    }
  }
}

// ---------------------------------------------------------------------------
extern "C" void kernel_launch(void* const* d_in, const int* in_sizes, int n_in,
                              void* d_out, int out_size, void* d_ws,
                              size_t ws_size, hipStream_t stream) {
  (void)in_sizes; (void)n_in; (void)out_size; (void)ws_size;
  const float* x      = (const float*)d_in[0];
  const float* g1     = (const float*)d_in[1];
  const float* b1     = (const float*)d_in[2];
  const float* qkv_w  = (const float*)d_in[3];
  const float* qkv_b  = (const float*)d_in[4];
  const float* rpb    = (const float*)d_in[5];
  const float* proj_w = (const float*)d_in[6];
  const float* proj_b = (const float*)d_in[7];
  const float* g2     = (const float*)d_in[8];
  const float* b2     = (const float*)d_in[9];
  const float* fc1_w  = (const float*)d_in[10];
  const float* fc1_b  = (const float*)d_in[11];
  const float* fc2_w  = (const float*)d_in[12];
  const float* fc2_b  = (const float*)d_in[13];
  float* out = (float*)d_out;
  char* ws   = (char*)d_ws;

  // workspace layout (bytes); total ~353 MB
  _Float16* qkvw16  = (_Float16*)(ws + 0);          // 576*192
  _Float16* projw16 = (_Float16*)(ws + 221184);     // 192*192
  _Float16* fc1w16  = (_Float16*)(ws + 294912);     // 768*192
  _Float16* fc2w16  = (_Float16*)(ws + 589824);     // 192*768
  _Float16* actbuf  = (_Float16*)(ws + 884736);     // [TOK,192] win / LN2 out
  _Float16* qkvbuf  = (_Float16*)(ws + 51216384);   // [TOK,576]
  _Float16* attnbuf = (_Float16*)(ws + 202211328);  // [TOK,192]
  _Float16* hidbuf  = qkvbuf;                       // [TOK,768] reuse qkv+attn
  float*    x1      = (float*)(ws + 252542976);     // [TOK,192] f32

  // weights -> f16 once per call (deterministic)
  cvt_f32_f16_kernel<<<(110592 + 255) / 256, 256, 0, stream>>>(qkv_w, qkvw16, 110592);
  cvt_f32_f16_kernel<<<(36864 + 255) / 256, 256, 0, stream>>>(proj_w, projw16, 36864);
  cvt_f32_f16_kernel<<<(147456 + 255) / 256, 256, 0, stream>>>(fc1_w, fc1w16, 147456);
  cvt_f32_f16_kernel<<<(147456 + 255) / 256, 256, 0, stream>>>(fc2_w, fc2w16, 147456);

  // LN1 + roll(-4,-4) + window partition -> f16
  ln_kernel<<<TOK / 4, 128, 0, stream>>>(x, g1, b1, actbuf, 1);

  // qkv GEMM: [TOK,192] x [576,192]^T   (64x64 macro-tiles)
  gemm64_kernel<0><<<(TOK / 64) * (QKV_DIM / 64) / 8, 256, 0, stream>>>(
      actbuf, qkvw16, qkv_b, qkvbuf, nullptr, nullptr, TOK, QKV_DIM, C_DIM);

  // fused windowed attention
  swin_attn_kernel<<<BWIN * NHEADS, 128, 0, stream>>>(qkvbuf, rpb, attnbuf);

  // proj GEMM + window-reverse + roll(+4,+4) + residual -> x1 (f32)
  gemm64_kernel<1><<<(TOK / 64) * (C_DIM / 64) / 8, 256, 0, stream>>>(
      attnbuf, projw16, proj_b, nullptr, x1, x, TOK, C_DIM, C_DIM);

  // LN2 -> f16
  ln_kernel<<<TOK / 4, 128, 0, stream>>>(x1, g2, b2, actbuf, 0);

  // fc1 GEMM + exact GELU -> f16
  gemm64_kernel<2><<<(TOK / 64) * (HID_DIM / 64) / 8, 256, 0, stream>>>(
      actbuf, fc1w16, fc1_b, hidbuf, nullptr, nullptr, TOK, HID_DIM, C_DIM);

  // fc2 GEMM + residual -> final f32 output
  gemm64_kernel<3><<<(TOK / 64) * (C_DIM / 64) / 8, 256, 0, stream>>>(
      hidbuf, fc2w16, fc2_b, nullptr, out, x1, TOK, C_DIM, HID_DIM);
}
